// CPC_16466904613092
// MI455X (gfx1250) — matrix operand order repaired and checked by reading, not compiled, not run on hardware
//
#include <hip/hip_runtime.h>
#include <hip/hip_bf16.h>
#include <math.h>

// ---------------------------------------------------------------------------
// CPC pipeline for MI455X (gfx1250), bf16 WMMA (v_wmma_f32_16x16x32_bf16)
// B-operands pre-packed into fragment-native layout -> b128 loads only.
// ---------------------------------------------------------------------------

typedef __bf16 bh;
typedef __attribute__((ext_vector_type(16))) __bf16 bhx16;
typedef __attribute__((ext_vector_type(8)))  __bf16 bhx8;
typedef __attribute__((ext_vector_type(8)))  float  f32x8;

#define DEV static __device__ __forceinline__

constexpr int Bb = 128, Tt = 256, Ww = 256, Dd = 128, Cc = 256, Kh = 12, Tp = 244;
constexpr float TEMP_INV = 10.0f;   // 1 / 0.1

DEV int lane_id() { return (int)(threadIdx.x & 31); }

// A-fragment: 16x32 bf16 tile, p -> element [0][0], row stride ld (elements).
// Lane layout (contiguous k, matches packed B): lanes 0-15 row M=lane, K in {0..15};
// lanes 16-31 row M=lane-16, K in {16..31}.
// Each lane's halves are contiguous -> two 16B vector loads.
DEV bhx16 load_a_bf16(const bh* __restrict__ p, int ld) {
  const int l = lane_id(), hs = l >> 4, m = l & 15;
  const bh* q = p + (size_t)m * ld + (hs << 4);
  bhx8 lo = *(const bhx8*)q;         // k = 16*hs + 0..7
  bhx8 hi = *(const bhx8*)(q + 8);   // k = 16*hs + 8..15
  bhx16 a;
#pragma unroll
  for (int e = 0; e < 8; ++e) { a[e] = lo[e]; a[8 + e] = hi[e]; }
  return a;
}

// Packed B tile: 512 bf16 in fragment order; lane reads 32 contiguous bytes.
DEV bhx16 load_b_packed(const bh* __restrict__ tile) {
  return *(const bhx16*)(tile + lane_id() * 16);
}

DEV f32x8 wmma_bf16(bhx16 a, bhx16 b, f32x8 c) {
  return __builtin_amdgcn_wmma_f32_16x16x32_bf16(false, a, false, b,
                                                 (short)0, c, false, false);
}

// D-fragment store: VGPR i holds row M=i+8*(lane/16), col N=lane%16.
DEV void store_d(float* p, int ld, f32x8 d) {
  const int l = lane_id(), hs = l >> 4, n = l & 15;
#pragma unroll
  for (int i = 0; i < 8; ++i) p[(i + 8 * hs) * ld + n] = d[i];
}

// Packed-layout index for B element (k, n) of a K x N matrix:
//   tile = (k/32)*(N/16) + n/16 ; lane = ((k>>4)&1)*16 + (n&15) ; e = k&15
DEV size_t pack_index(int k, int n, int N) {
  int tile = (k >> 5) * (N >> 4) + (n >> 4);
  int lane = (((k >> 4) & 1) << 4) | (n & 15);
  return ((size_t)tile << 9) + (lane << 4) + (k & 15);
}

// ---------------------------------------------------------------------------
// Pack an f32 weight matrix into fragment-native bf16 B layout.
// transpose==0: src is K x N row-major.  transpose==1: src is N x K row-major.
// ---------------------------------------------------------------------------
__global__ void k_pack_b(const float* __restrict__ src, bh* __restrict__ dst,
                         int K, int N, int transpose) {
  int total = K * N;
  int Ntiles = N >> 4;
  for (int i = blockIdx.x * blockDim.x + threadIdx.x; i < total;
       i += gridDim.x * blockDim.x) {
    int tile = i >> 9, rem = i & 511;
    int lane = rem >> 4, e = rem & 15;
    int kt = tile / Ntiles, nt = tile - kt * Ntiles;
    int k = kt * 32 + ((lane >> 4) << 4) + e;
    int n = nt * 16 + (lane & 15);
    float v = transpose ? src[(size_t)n * K + k] : src[(size_t)k * N + n];
    dst[i] = (bh)v;
  }
}

__global__ void k_zero(float* __restrict__ hF, bh* __restrict__ hB,
                       float* __restrict__ accum) {
  int i = blockIdx.x * blockDim.x + threadIdx.x;
  if (i < Bb * Cc) { hF[i] = 0.f; hB[i] = (bh)0.f; }
  if (i < 2) accum[i] = 0.f;
}

// ---------------------------------------------------------------------------
// Encoder: z = l2norm(relu(X@W1+b1)@W2+b2).  One wave = 16 (b,t) rows.
// Writes z f32 (d_out), z bf16 row-major (A for GRU/heads),
// and z_t in PACKED B layout per batch (for the logits GEMM).
// ---------------------------------------------------------------------------
__global__ void k_encoder(const float* __restrict__ rr, const bh* __restrict__ w1,
                          const float* __restrict__ b1, const bh* __restrict__ w2,
                          const float* __restrict__ b2,
                          float* __restrict__ z_out, bh* __restrict__ z_bf,
                          bh* __restrict__ z_t) {
  extern __shared__ char smem[];
  const int wave = (int)(threadIdx.x >> 5);
  const int l = lane_id(), hs = l >> 4, nl = l & 15;
  bh* Xs = (bh*)smem + (size_t)wave * (16 * Ww);
  bh* Hs = (bh*)smem + (size_t)4 * (16 * Ww) + (size_t)wave * (16 * 256);
  const int row0 = (blockIdx.x * 4 + wave) * 16;

  // Load + convert the contiguous 16x256 f32 input tile.
  {
    const float* src = rr + (size_t)row0 * Ww;
    for (int i = l * 4; i < 16 * Ww; i += 32 * 4) {
      float4 v = *(const float4*)(src + i);
      Xs[i + 0] = (bh)v.x; Xs[i + 1] = (bh)v.y;
      Xs[i + 2] = (bh)v.z; Xs[i + 3] = (bh)v.w;
    }
  }
  __syncthreads();

  // GEMM1 (16x256 @ 256x256) + bias + relu -> Hs (bf16).  w1 packed: Ntiles=16.
  for (int nt = 0; nt < 16; ++nt) {
    f32x8 acc = {};
    for (int kt = 0; kt < 8; ++kt) {
      bhx16 a = load_a_bf16(Xs + kt * 32, Ww);
      bhx16 b = load_b_packed(w1 + (((size_t)kt * 16 + nt) << 9));
      acc = wmma_bf16(a, b, acc);
    }
    float bias = b1[nt * 16 + nl];
#pragma unroll
    for (int i = 0; i < 8; ++i) {
      float v = acc[i] + bias;
      v = v > 0.f ? v : 0.f;
      Hs[(i + 8 * hs) * 256 + nt * 16 + nl] = (bh)v;
    }
  }
  __syncthreads();

  // GEMM2 (16x256 @ 256x128).  w2 packed: Ntiles=8.
  f32x8 z[8];
  for (int nt = 0; nt < 8; ++nt) {
    f32x8 acc = {};
    for (int kt = 0; kt < 8; ++kt) {
      bhx16 a = load_a_bf16(Hs + kt * 32, 256);
      bhx16 b = load_b_packed(w2 + (((size_t)kt * 8 + nt) << 9));
      acc = wmma_bf16(a, b, acc);
    }
    float bias = b2[nt * 16 + nl];
#pragma unroll
    for (int i = 0; i < 8; ++i) acc[i] += bias;
    z[nt] = acc;
  }

  // Row L2 norm: row m=i+8*hs lives in a 16-lane half -> xor masks 8,4,2,1.
  float sc[8];
#pragma unroll
  for (int i = 0; i < 8; ++i) {
    float s = 0.f;
#pragma unroll
    for (int nt = 0; nt < 8; ++nt) s += z[nt][i] * z[nt][i];
    for (int m = 8; m; m >>= 1) s += __shfl_xor(s, m, 32);
    sc[i] = 1.0f / fmaxf(sqrtf(s), 1e-12f);
  }

  for (int nt = 0; nt < 8; ++nt) {
#pragma unroll
    for (int i = 0; i < 8; ++i) {
      int bt  = row0 + i + 8 * hs;
      int col = nt * 16 + nl;            // d index
      float v = z[nt][i] * sc[i];
      z_out[(size_t)bt * Dd + col] = v;
      z_bf [(size_t)bt * Dd + col] = (bh)v;
      int bi = bt / Tt, ti = bt - bi * Tt;
      // z_t: per-batch packed B matrix, K = D = 128, N = T = 256
      z_t[(size_t)bi * (Dd * Tt) + pack_index(col, ti, Tt)] = (bh)v;
    }
  }
}

// ---------------------------------------------------------------------------
// GRU step t: gh/gx GEMMs (3 gates) fused with the pointwise update.
// Block = 4 waves; wave covers a 16(b) x 16(c) tile for all 3 gates.
// wihT packed: K=128, N=768 (Ntiles=48).  whhT packed: K=256, N=768.
// ---------------------------------------------------------------------------
__global__ void k_gru_step(const bh* __restrict__ z_bf, const bh* __restrict__ wihT,
                           const bh* __restrict__ whhT, const float* __restrict__ b_ih,
                           const float* __restrict__ b_hh,
                           const float* __restrict__ h_in, const bh* __restrict__ h_in_bf,
                           float* __restrict__ h_out, bh* __restrict__ h_out_bf,
                           float* __restrict__ c_out, bh* __restrict__ c_out_bf, int t) {
  extern __shared__ char smem[];
  const int wave = (int)(threadIdx.x >> 5);
  const int l = lane_id();
  float* tiles = (float*)smem + (size_t)wave * 6 * 256;
  const int b0 = blockIdx.x * 16;
  const int c0 = blockIdx.y * 64 + wave * 16;
  const int ct = c0 >> 4;                 // column-tile index within a gate

  f32x8 xacc[3] = {};
  f32x8 hacc[3] = {};

  const bh* abase_z = z_bf + ((size_t)b0 * Tt + t) * Dd;
  const bh* abase_h = h_in_bf + (size_t)b0 * Cc;

  // x-part: K = D = 128
  for (int kt = 0; kt < 4; ++kt) {
    bhx16 a = load_a_bf16(abase_z + kt * 32, Tt * Dd);
#pragma unroll
    for (int g = 0; g < 3; ++g) {
      const bh* tile = wihT + (((size_t)kt * 48 + g * 16 + ct) << 9);
      __builtin_prefetch(tile + 512, 0, 1);
      xacc[g] = wmma_bf16(a, load_b_packed(tile), xacc[g]);
    }
  }
  // h-part: K = C = 256
  for (int kt = 0; kt < 8; ++kt) {
    bhx16 a = load_a_bf16(abase_h + kt * 32, Cc);
#pragma unroll
    for (int g = 0; g < 3; ++g) {
      const bh* tile = whhT + (((size_t)kt * 48 + g * 16 + ct) << 9);
      __builtin_prefetch(tile + 512, 0, 1);
      hacc[g] = wmma_bf16(a, load_b_packed(tile), hacc[g]);
    }
  }
#pragma unroll
  for (int g = 0; g < 3; ++g) {
    store_d(tiles + g * 256, 16, xacc[g]);
    store_d(tiles + (3 + g) * 256, 16, hacc[g]);
  }
  __syncthreads();

  for (int j = 0; j < 8; ++j) {
    int e = j * 32 + l;
    int m = e >> 4, cl = e & 15;
    int bi = b0 + m, c = c0 + cl;
    float xr = tiles[0 * 256 + m * 16 + cl] + b_ih[c];
    float xz = tiles[1 * 256 + m * 16 + cl] + b_ih[256 + c];
    float xn = tiles[2 * 256 + m * 16 + cl] + b_ih[512 + c];
    float hr = tiles[3 * 256 + m * 16 + cl] + b_hh[c];
    float hz = tiles[4 * 256 + m * 16 + cl] + b_hh[256 + c];
    float hn = tiles[5 * 256 + m * 16 + cl] + b_hh[512 + c];
    float r  = 1.f / (1.f + __expf(-(xr + hr)));
    float zg = 1.f / (1.f + __expf(-(xz + hz)));
    float n  = tanhf(xn + r * hn);
    float hp = h_in[(size_t)bi * Cc + c];
    float hv = (1.f - zg) * n + zg * hp;
    h_out   [(size_t)bi * Cc + c] = hv;
    h_out_bf[(size_t)bi * Cc + c] = (bh)hv;
    size_t ci = ((size_t)bi * Tt + t) * Cc + c;
    c_out[ci]    = hv;
    c_out_bf[ci] = (bh)hv;
  }
}

// ---------------------------------------------------------------------------
// Per (k,b) block: preds = l2norm(c_used @ Wk + bk)  (in LDS),
// logits = preds @ z_t[b] * (1/TEMP)  (per-wave LDS tile),
// then logsumexp / pos / neg-max reductions -> atomic loss/accuracy sums.
// wk packed per head: K=256, N=128 (Ktiles=8, Ntiles=8, stride 32768 elems).
// z_t packed per batch: K=128, N=256 (Ktiles=4, Ntiles=16, stride 32768 elems).
// ---------------------------------------------------------------------------
__global__ void k_pred_logits_loss(const bh* __restrict__ c_bf, const bh* __restrict__ wk,
                                   const float* __restrict__ wk_b,
                                   const bh* __restrict__ z_t,
                                   float* __restrict__ accum) {
  extern __shared__ char smem[];
  float* bufF = (float*)smem;                 // 32768 f32 (pred pre-norm, then logits)
  bh* predB   = (bh*)(smem + 131072);         // 32768 bf16 (normalized preds)
  const int wave = (int)(threadIdx.x >> 5);
  const int l = lane_id(), hs = l >> 4, nl = l & 15;
  const int kk = blockIdx.x, b = blockIdx.y;
  const bh* wk_h = wk + (size_t)kk * 32768;
  const bh* zt_b = z_t + (size_t)b * 32768;

  // Phase 1: pred pre-norm tiles (16 m-tiles x 8 n-tiles, K=256)
  for (int tid = wave; tid < 128; tid += 8) {
    int mt = tid >> 3, nt = tid & 7;
    f32x8 acc = {};
    for (int kt = 0; kt < 8; ++kt) {
      bhx16 a  = load_a_bf16(c_bf + ((size_t)b * Tt + mt * 16) * Cc + kt * 32, Cc);
      const bh* tile = wk_h + (((size_t)kt * 8 + nt) << 9);
      __builtin_prefetch(tile + 512, 0, 1);
      acc = wmma_bf16(a, load_b_packed(tile), acc);
    }
    float bias = wk_b[kk * Dd + nt * 16 + nl];
#pragma unroll
    for (int i = 0; i < 8; ++i)
      bufF[(mt * 16 + i + 8 * hs) * Dd + nt * 16 + nl] = acc[i] + bias;
  }
  __syncthreads();

  // Phase 2: row l2norm -> predB (bf16)
  for (int r = wave; r < 256; r += 8) {
    float v[4], s = 0.f;
#pragma unroll
    for (int j = 0; j < 4; ++j) { v[j] = bufF[r * Dd + l * 4 + j]; s += v[j] * v[j]; }
    for (int m = 16; m; m >>= 1) s += __shfl_xor(s, m, 32);
    float sc = 1.f / fmaxf(sqrtf(s), 1e-12f);
#pragma unroll
    for (int j = 0; j < 4; ++j) predB[r * Dd + l * 4 + j] = (bh)(v[j] * sc);
  }
  __syncthreads();

  // Phase 3: logits tile per wave (bufF reused as 8 x 16KB wave buffers)
  float* myLog = bufF + (size_t)wave * 16 * 256;
  float lossAcc = 0.f, hitAcc = 0.f;
  for (int mt = wave; mt < 16; mt += 8) {
    for (int st = 0; st < 16; ++st) {
      f32x8 acc = {};
      for (int kt = 0; kt < 4; ++kt) {
        bhx16 a = load_a_bf16(predB + (size_t)(mt * 16) * Dd + kt * 32, Dd);
        acc = wmma_bf16(a, load_b_packed(zt_b + (((size_t)kt * 16 + st) << 9)), acc);
      }
#pragma unroll
      for (int i = 0; i < 8; ++i) acc[i] *= TEMP_INV;
      store_d(myLog + st * 16, 256, acc);
    }
    for (int r = 0; r < 16; ++r) {
      int t = mt * 16 + r;
      if (t >= Tp) continue;
      int pos = t + kk + 1;
      float vals[8], mAll = -INFINITY, mNeg = -INFINITY;
#pragma unroll
      for (int j = 0; j < 8; ++j) {
        int c = l * 8 + j;
        float v = myLog[r * 256 + c];
        vals[j] = v;
        mAll = fmaxf(mAll, v);
        if (c != pos) mNeg = fmaxf(mNeg, v);
      }
      for (int m = 16; m; m >>= 1) {
        mAll = fmaxf(mAll, __shfl_xor(mAll, m, 32));
        mNeg = fmaxf(mNeg, __shfl_xor(mNeg, m, 32));
      }
      float s = 0.f;
#pragma unroll
      for (int j = 0; j < 8; ++j) s += __expf(vals[j] - mAll);
      for (int m = 16; m; m >>= 1) s += __shfl_xor(s, m, 32);
      float posv = myLog[r * 256 + pos];
      lossAcc += (mAll + __logf(s)) - posv;
      hitAcc  += (posv >= mNeg) ? 1.f : 0.f;
    }
  }
  if (l == 0) {
    atomicAdd(&accum[0], lossAcc);
    atomicAdd(&accum[1], hitAcc);
  }
}

__global__ void k_finalize(const float* __restrict__ accum, float* __restrict__ out) {
  if (threadIdx.x == 0) {
    const float invN = 1.0f / (float)(Kh * Bb * Tp);
    out[0] = accum[0] * invN;
    out[1] = accum[1] * invN * 100.0f;
  }
}

// ---------------------------------------------------------------------------
extern "C" void kernel_launch(void* const* d_in, const int* in_sizes, int n_in,
                              void* d_out, int out_size, void* d_ws, size_t ws_size,
                              hipStream_t stream) {
  (void)in_sizes; (void)n_in; (void)out_size; (void)ws_size;
  const float* rr  = (const float*)d_in[0];
  const float* w1  = (const float*)d_in[1];
  const float* b1  = (const float*)d_in[2];
  const float* w2  = (const float*)d_in[3];
  const float* b2  = (const float*)d_in[4];
  const float* Wih = (const float*)d_in[5];
  const float* Whh = (const float*)d_in[6];
  const float* bih = (const float*)d_in[7];
  const float* bhh = (const float*)d_in[8];
  const float* Wkw = (const float*)d_in[9];
  const float* Wkb = (const float*)d_in[10];

  float* out   = (float*)d_out;
  float* z_out = out + 2;
  float* c_out = out + 2 + (size_t)Bb * Tt * Dd;

  char* ws = (char*)d_ws;
  size_t off = 0;
  auto alloc = [&](size_t bytes) -> char* {
    char* p = ws + off;
    off += (bytes + 255) & ~(size_t)255;
    return p;
  };
  bh* w1_bf = (bh*)alloc((size_t)256 * 256 * 2);
  bh* w2_bf = (bh*)alloc((size_t)256 * 128 * 2);
  bh* wihT  = (bh*)alloc((size_t)128 * 768 * 2);
  bh* whhT  = (bh*)alloc((size_t)256 * 768 * 2);
  bh* wk_bf = (bh*)alloc((size_t)12 * 256 * 128 * 2);
  bh* z_bf  = (bh*)alloc((size_t)Bb * Tt * Dd * 2);
  bh* z_t   = (bh*)alloc((size_t)Bb * Dd * Tt * 2);
  bh* c_bf  = (bh*)alloc((size_t)Bb * Tt * Cc * 2);
  float* hF0 = (float*)alloc((size_t)Bb * Cc * 4);
  float* hF1 = (float*)alloc((size_t)Bb * Cc * 4);
  bh* hB0 = (bh*)alloc((size_t)Bb * Cc * 2);
  bh* hB1 = (bh*)alloc((size_t)Bb * Cc * 2);
  float* accum = (float*)alloc(256);

  // Pack weights into fragment-native bf16 B layout.
  // (B matrix =) w1: 256x256; w2: 256x128; W_ih^T: 128x768 (src transposed);
  // W_hh^T: 256x768 (src transposed); Wk[k]: 256x128 each.
  k_pack_b<<<dim3(64),  dim3(256), 0, stream>>>(w1,  w1_bf, 256, 256, 0);
  k_pack_b<<<dim3(32),  dim3(256), 0, stream>>>(w2,  w2_bf, 256, 128, 0);
  k_pack_b<<<dim3(96),  dim3(256), 0, stream>>>(Wih, wihT, 128, 768, 1);
  k_pack_b<<<dim3(192), dim3(256), 0, stream>>>(Whh, whhT, 256, 768, 1);
  for (int k = 0; k < Kh; ++k)
    k_pack_b<<<dim3(32), dim3(256), 0, stream>>>(Wkw + (size_t)k * 256 * 128,
                                                 wk_bf + (size_t)k * 32768,
                                                 256, 128, 0);
  k_zero<<<dim3((Bb * Cc + 255) / 256), dim3(256), 0, stream>>>(hF0, hB0, accum);

  // Encoder: 512 blocks x 4 waves x 16 rows = 32768 rows, 64KB LDS/block
  k_encoder<<<dim3(512), dim3(128), 65536, stream>>>(rr, w1_bf, b1, w2_bf, b2,
                                                     z_out, z_bf, z_t);

  // GRU recurrence: 256 stream-ordered step kernels, ping-pong h buffers
  for (int t = 0; t < Tt; ++t) {
    float* hi = (t & 1) ? hF1 : hF0;
    float* ho = (t & 1) ? hF0 : hF1;
    bh* hbi = (t & 1) ? hB1 : hB0;
    bh* hbo = (t & 1) ? hB0 : hB1;
    k_gru_step<<<dim3(8, 4), dim3(128), 24576, stream>>>(
        z_bf, wihT, whhT, bih, bhh, hi, hbi, ho, hbo, c_out, c_bf, t);
  }

  // Fused preds + logits + loss: one block per (k,b), 192KB LDS
  k_pred_logits_loss<<<dim3(12, 128), dim3(256), 196608, stream>>>(
      c_bf, wk_bf, Wkb, z_t, accum);
  k_finalize<<<dim3(1), dim3(1), 0, stream>>>(accum, out);
}
